// TreeAttention_91190745628782
// MI455X (gfx1250) — matrix operand order repaired and checked by reading, not compiled, not run on hardware
//
#include <hip/hip_runtime.h>
#include <hip/hip_bf16.h>

// MI455X / gfx1250 tree-attention pipeline, bf16 WMMA compute, fp32 accumulate.
// Double-buffered LDS with async (ASYNCcnt) and TDM (TENSORcnt) tile movers so
// global->LDS copies overlap the WMMA work.
// Workspace layout (requires ws_size >= 48 MiB):
//   [ 0, 8M)  X bf16            [4096 x 1024]
//   [ 8,10M)  Wq bf16   [10,12M) Wk   [12,14M) Wv   [14,16M) Wo   (1024x1024)
//   [16,24M)  Q bf16            [b, h, n, d] = [2,16,2048,64] (pre-scaled d^-1/2)
//   [24,32M)  K bf16   [32,40M) V bf16
//   [40,48M)  ctx bf16          [4096 x 1024]

#define USE_ASYNC_LDS 1   // global_load_async_to_lds_b128 for the GEMM A tile
#define USE_TDM       1   // tensor_load_to_lds for the attention K tile

typedef __bf16 bf16;
typedef __attribute__((ext_vector_type(16))) __bf16 bf16x16;
typedef __attribute__((ext_vector_type(8)))  float  f32x8;
typedef __attribute__((ext_vector_type(4)))  unsigned int u32x4;
typedef __attribute__((ext_vector_type(4)))  int  i32x4;
typedef __attribute__((ext_vector_type(8)))  int  i32x8;

union FragU { bf16x16 v; uint4 u[2]; bf16 h[16]; };
union U2H4  { uint2 u; bf16 h[4]; };
union H2U   { bf16 h[2]; unsigned u; };
union I4U   { int4 v; int i[4]; };

#define WMMA_BF16(A, B, C) \
  __builtin_amdgcn_wmma_f32_16x16x32_bf16(false, (A), false, (B), (short)0, (C), false, false)

#define NEG_MAX_F32 (-3.4028235e38f)

// LDS byte offset of a __shared__ object: flat shared addresses carry the LDS
// offset in the low 32 bits (ISA 10.2: LDS aperture truncates to addr[31:0]).
__device__ __forceinline__ unsigned lds_off(const void* p) {
  return (unsigned)(size_t)p;
}

#if USE_ASYNC_LDS
__device__ __forceinline__ void async_load_b128(unsigned lds, unsigned long long gaddr) {
  asm volatile("global_load_async_to_lds_b128 %0, %1, off"
               :: "v"(lds), "v"(gaddr) : "memory");
}
__device__ __forceinline__ void async_wait0() {
  asm volatile("s_wait_asynccnt 0" ::: "memory");
}
#endif

#if USE_TDM
// 1-D contiguous bf16 tile copy via the Tensor Data Mover (D# per ISA 8.3/8.4).
__device__ __forceinline__ void tdm_load_tile_bf16(unsigned ldsByteOff,
                                                   const bf16* gsrc,
                                                   unsigned nElems) {
  unsigned long long ga = (unsigned long long)(size_t)gsrc;
  u32x4 g0;
  g0[0] = 1u;                                   // count=1, is_restore=0, gather=0
  g0[1] = ldsByteOff;                           // lds_addr
  g0[2] = (unsigned)ga;                         // global_addr[31:0]
  g0[3] = (unsigned)(ga >> 32) | (2u << 30);    // global_addr[56:32] | type=2
  i32x8 g1;
  g1[0] = 1 << 16;                              // data_size = 2 bytes; wg_mask=0
  g1[1] = (int)((nElems & 0xFFFFu) << 16);      // tensor_dim0[15:0]
  g1[2] = (int)((nElems >> 16) | (1u << 16));   // tensor_dim0[31:16], tensor_dim1=1
  g1[3] = (int)((nElems & 0xFFFFu) << 16);      // tile_dim0
  g1[4] = 1;                                    // tile_dim1 = 1, tile_dim2 = 0
  g1[5] = (int)nElems;                          // tensor_dim0_stride[31:0]
  g1[6] = 0;
  g1[7] = 0;
  i32x4 z4 = {0, 0, 0, 0};
  i32x8 z8 = {0, 0, 0, 0, 0, 0, 0, 0};
#if defined(__clang_major__) && (__clang_major__ >= 23)
  __builtin_amdgcn_tensor_load_to_lds(g0, g1, z4, z4, z8, 0);
#else
  __builtin_amdgcn_tensor_load_to_lds(g0, g1, z4, z4, 0);
  (void)z8;
#endif
}
#endif

// ---------------------------------------------------------------- cast ------
__global__ void cast_f32_bf16(const float* __restrict__ in, bf16* __restrict__ out, int n4) {
  int i = blockIdx.x * blockDim.x + threadIdx.x;
  if (i < n4) {
    float4 f = ((const float4*)in)[i];
    union { bf16 h[4]; uint2 u; } p;
    p.h[0] = (bf16)f.x; p.h[1] = (bf16)f.y; p.h[2] = (bf16)f.z; p.h[3] = (bf16)f.w;
    ((uint2*)out)[i] = p.u;
  }
}

// ------------------------------------------------- shared GEMM tile core ----
// Block tile 128(M) x 64(N), K = 1024 in steps of 32. 8 waves, each wave owns a
// 32x32 subtile = 2x2 WMMA tiles. Double-buffered LDS: A row-major [128][32]
// (filled by per-lane async global->LDS b128), B transposed [64 n][32 k]
// (VGPR transpose, k-paired ds_store_b32).
__device__ __forceinline__ void gemm_stage(const bf16* __restrict__ X,
                                           const bf16* __restrict__ W,
                                           int m0, int n0, int k0,
                                           unsigned sAoff, bf16* sBt, int t) {
  // A tile: 128 rows x 32 k = 512 x b128, 2 async loads per thread.
  #pragma unroll
  for (int j = 0; j < 2; ++j) {
    int u = t + 256 * j;
    int row = u >> 2, kq = u & 3;
#if USE_ASYNC_LDS
    unsigned long long ga =
        (unsigned long long)(size_t)(X + (size_t)(m0 + row) * 1024 + k0) +
        (unsigned long long)kq * 16ull;
    async_load_b128(sAoff + (unsigned)u * 16u, ga);
#else
    ((uint4*)(size_t)0)[0] = ((uint4*)(size_t)0)[0];  // unreachable placeholder
#endif
  }
#if !USE_ASYNC_LDS
  {
    uint4* sA4 = (uint4*)(size_t)sAoff;  // not used in this configuration
    (void)sA4;
  }
#endif
  // B tile: 32 k x 64 n -> sBt[n][k], two rows (k, k+1) packed per b32 store.
  {
    int kp = (t >> 4) * 2, nb = (t & 15) * 4;
    U2H4 w0, w1;
    w0.u = *(const uint2*)(W + (size_t)(k0 + kp) * 1024 + n0 + nb);
    w1.u = *(const uint2*)(W + (size_t)(k0 + kp + 1) * 1024 + n0 + nb);
    #pragma unroll
    for (int e = 0; e < 4; ++e) {
      H2U p; p.h[0] = w0.h[e]; p.h[1] = w1.h[e];
      *(unsigned*)(sBt + (nb + e) * 32 + kp) = p.u;
    }
  }
}

__device__ __forceinline__ void gemm_compute(const bf16* sA, const bf16* sBt,
                                             int wm, int wn, int hh, int ln,
                                             f32x8 (&acc)[2][2]) {
  const uint4* sA4  = (const uint4*)sA;
  const uint4* sBt4 = (const uint4*)sBt;
  FragU a[2], bfr[2];
  #pragma unroll
  for (int i = 0; i < 2; ++i) {            // A 16x32 bf16 layout (ISA 7.12.2)
    int row = wm + i * 16 + ln;            // lane<16: K {0..7,16..23}; else {8..15,24..31}
    a[i].u[0] = sA4[row * 4 + hh];
    a[i].u[1] = sA4[row * 4 + 2 + hh];
  }
  #pragma unroll
  for (int j = 0; j < 2; ++j) {            // B 32x16: lane<16 K {0..7,16..23}; else {8..15,24..31}
    int col = wn + j * 16 + ln;
    bfr[j].u[0] = sBt4[col * 4 + hh];
    bfr[j].u[1] = sBt4[col * 4 + 2 + hh];
  }
  #pragma unroll
  for (int i = 0; i < 2; ++i)
    #pragma unroll
    for (int j = 0; j < 2; ++j)
      acc[i][j] = WMMA_BF16(a[i].v, bfr[j].v, acc[i][j]);
}

__device__ __forceinline__ void gemm_tile_128x64(
    const bf16* __restrict__ X, const bf16* __restrict__ W,
    int m0, int n0, bf16 (&sA)[2][128 * 32], bf16 (&sBt)[2][64 * 32],
    f32x8 (&acc)[2][2]) {
  const int t    = threadIdx.x;
  const int wave = t >> 5, lane = t & 31;
  const int hh   = lane >> 4, ln = lane & 15;
  const int wm   = (wave >> 1) * 32, wn = (wave & 1) * 32;
  const unsigned sAoff[2] = { lds_off(sA[0]), lds_off(sA[1]) };

  // Prologue: stage k0 = 0 into buffer 0.
  gemm_stage(X, W, m0, n0, 0, sAoff[0], sBt[0], t);
#if USE_ASYNC_LDS
  async_wait0();
#endif
  __syncthreads();

  for (int it = 0; it < 32; ++it) {
    const int cur = it & 1, nxt = cur ^ 1;
    // Issue next tile's copies first; they run under the WMMAs below.
    if (it + 1 < 32)
      gemm_stage(X, W, m0, n0, (it + 1) * 32, sAoff[nxt], sBt[nxt], t);
    gemm_compute(sA[cur], sBt[cur], wm, wn, hh, ln, acc);
#if USE_ASYNC_LDS
    async_wait0();       // next buffer's async loads complete
#endif
    __syncthreads();     // publishes next buffer; retires current reads
  }
}

// --------------------------------------------------------- QKV projection ---
__global__ void qkv_gemm(const bf16* __restrict__ X,
                         const bf16* __restrict__ Wq, const bf16* __restrict__ Wk,
                         const bf16* __restrict__ Wv,
                         const float* __restrict__ bq, const float* __restrict__ bk,
                         const float* __restrict__ bv,
                         bf16* __restrict__ Q, bf16* __restrict__ K, bf16* __restrict__ V) {
  __shared__ bf16 sA[2][128 * 32];
  __shared__ bf16 sBt[2][64 * 32];
  const int m0 = blockIdx.x * 128, n0 = blockIdx.y * 64;
  const bf16*  W    = (blockIdx.z == 0) ? Wq : (blockIdx.z == 1) ? Wk : Wv;
  const float* bias = (blockIdx.z == 0) ? bq : (blockIdx.z == 1) ? bk : bv;
  bf16*        Out  = (blockIdx.z == 0) ? Q  : (blockIdx.z == 1) ? K  : V;
  const float scale = (blockIdx.z == 0) ? 0.125f : 1.0f;  // fold d^-1/2 into Q

  f32x8 acc[2][2];
  #pragma unroll
  for (int i = 0; i < 2; ++i)
    #pragma unroll
    for (int j = 0; j < 2; ++j)
      #pragma unroll
      for (int r = 0; r < 8; ++r) acc[i][j][r] = 0.f;

  gemm_tile_128x64(X, W, m0, n0, sA, sBt, acc);

  const int t = threadIdx.x, wave = t >> 5, lane = t & 31;
  const int hh = lane >> 4, ln = lane & 15;
  #pragma unroll
  for (int i = 0; i < 2; ++i)
    #pragma unroll
    for (int j = 0; j < 2; ++j) {
      int n = n0 + (wave & 1) * 32 + j * 16 + ln;
      int head = n >> 6, d = n & 63;
      float bs = bias[n];
      #pragma unroll
      for (int r = 0; r < 8; ++r) {
        int m   = m0 + (wave >> 1) * 32 + i * 16 + r + 8 * hh;
        int btk = m >> 11, tok = m & 2047;
        float val = (acc[i][j][r] + bs) * scale;
        Out[(((size_t)(btk * 16 + head)) * 2048 + tok) * 64 + d] = (bf16)val;
      }
    }
}

// --------------------------------------------------------- out projection ---
__global__ void out_gemm(const bf16* __restrict__ Ctx, const bf16* __restrict__ Wo,
                         const float* __restrict__ bo, float* __restrict__ Outp) {
  __shared__ bf16 sA[2][128 * 32];
  __shared__ bf16 sBt[2][64 * 32];
  const int m0 = blockIdx.x * 128, n0 = blockIdx.y * 64;

  f32x8 acc[2][2];
  #pragma unroll
  for (int i = 0; i < 2; ++i)
    #pragma unroll
    for (int j = 0; j < 2; ++j)
      #pragma unroll
      for (int r = 0; r < 8; ++r) acc[i][j][r] = 0.f;

  gemm_tile_128x64(Ctx, Wo, m0, n0, sA, sBt, acc);

  const int t = threadIdx.x, wave = t >> 5, lane = t & 31;
  const int hh = lane >> 4, ln = lane & 15;
  #pragma unroll
  for (int i = 0; i < 2; ++i)
    #pragma unroll
    for (int j = 0; j < 2; ++j) {
      int n = n0 + (wave & 1) * 32 + j * 16 + ln;
      float bs = bo[n];
      #pragma unroll
      for (int r = 0; r < 8; ++r) {
        int m = m0 + (wave >> 1) * 32 + i * 16 + r + 8 * hh;
        Outp[(size_t)m * 1024 + n] = acc[i][j][r] + bs;
      }
    }
}

// ------------------------------------------------------------- attention ----
// One wave = one 16-query tile of one (b,h); 8 waves share double-buffered LDS
// K/V tiles. Scores computed transposed (S' = K Q^T): per-lane mask words are
// contiguous along keys, and each lane's exp(P) values are exactly its bf16
// A-fragment for the P*V WMMA (no LDS round-trip). The contiguous 4 KB K tile
// is fetched by a single TDM tensor_load_to_lds, double-buffered so the DMA
// for tile i+1 runs under the compute of tile i.
__global__ void tree_attn(const bf16* __restrict__ Q, const bf16* __restrict__ K,
                          const bf16* __restrict__ V, const int* __restrict__ mask,
                          bf16* __restrict__ Ctx) {
  __shared__ bf16 sK[2][32 * 64];   // K tile, row-major [key][d]
  __shared__ bf16 sVt[2][64 * 32];  // V tile transposed [d][key]
  const int t = threadIdx.x, wave = t >> 5, lane = t & 31;
  const int hh = lane >> 4, ln = lane & 15;
  const int h = blockIdx.y, bb = blockIdx.z;
  const int q0 = blockIdx.x * 128 + wave * 16;
  const size_t headBase = ((size_t)(bb * 16 + h)) * 2048 * 64;
#if USE_TDM
  const unsigned sKoff[2] = { lds_off(sK[0]), lds_off(sK[1]) };
#endif

  // Q^T B-fragments (d-chunks of 32): lane holds column q = q0+ln, 16 contig d.
  FragU qf[2];
  {
    const bf16* qrow = Q + headBase + (size_t)(q0 + ln) * 64;
    #pragma unroll
    for (int j = 0; j < 2; ++j) {
      qf[j].u[0] = ((const uint4*)(qrow + 32 * j + hh * 16))[0];
      qf[j].u[1] = ((const uint4*)(qrow + 32 * j + hh * 16))[1];
    }
  }

  f32x8 o[4];
  #pragma unroll
  for (int c = 0; c < 4; ++c)
    #pragma unroll
    for (int r = 0; r < 8; ++r) o[c][r] = 0.f;
  float rowMax = -INFINITY, rowSum = 0.f;

  // Stage helper (K via TDM from wave 0; V transposed with k-paired b32 stores).
  auto stage = [&](int kt, int buf) {
#if USE_TDM
    if (wave == 0)
      tdm_load_tile_bf16(sKoff[buf], K + headBase + (size_t)kt * 64, 32 * 64);
#else
    {
      uint4* d = (uint4*)sK[buf];
      int row = t >> 3, part = t & 7;
      d[t] = ((const uint4*)(K + headBase + (size_t)(kt + row) * 64))[part];
    }
#endif
    {
      int kp = (t >> 4) * 2, db = (t & 15) * 4;
      U2H4 v0, v1;
      v0.u = *(const uint2*)(V + headBase + (size_t)(kt + kp) * 64 + db);
      v1.u = *(const uint2*)(V + headBase + (size_t)(kt + kp + 1) * 64 + db);
      #pragma unroll
      for (int e = 0; e < 4; ++e) {
        H2U p; p.h[0] = v0.h[e]; p.h[1] = v1.h[e];
        *(unsigned*)(sVt[buf] + (db + e) * 32 + kp) = p.u;
      }
    }
  };

  // Prologue: stage tile 0 into buffer 0.
  stage(0, 0);
#if USE_TDM
  if (wave == 0) __builtin_amdgcn_s_wait_tensorcnt((short)0);
#endif
  __syncthreads();

  for (int it = 0; it < 64; ++it) {
    const int cur = it & 1, nxt = cur ^ 1;
    const int kt = it * 32;
    // Kick off next tile's DMA + V staging; runs under the compute below.
    if (it + 1 < 64) stage(kt + 32, nxt);

    const uint4* sK4  = (const uint4*)sK[cur];
    const uint4* sVt4 = (const uint4*)sVt[cur];

    // S' = K * Q^T : rows = keys, cols = queries. Two 16-key accumulators.
    f32x8 s[2];
    #pragma unroll
    for (int i = 0; i < 2; ++i)
      #pragma unroll
      for (int r = 0; r < 8; ++r) s[i][r] = 0.f;
    #pragma unroll
    for (int i = 0; i < 2; ++i)
      #pragma unroll
      for (int j = 0; j < 2; ++j) {
        FragU a;
        int row = i * 16 + ln;
        a.u[0] = sK4[row * 8 + 4 * j + 2 * hh];
        a.u[1] = sK4[row * 8 + 4 * j + 2 * hh + 1];
        s[i] = WMMA_BF16(a.v, qf[j].v, s[i]);
      }

    // Tree mask: lane's 8 accumulator slots are 8 consecutive keys.
    const int qg = q0 + ln;
    #pragma unroll
    for (int i = 0; i < 2; ++i) {
      const int* mrow = mask + (size_t)qg * 2048 + kt + i * 16 + hh * 8;
      I4U m0v, m1v;
      m0v.v = ((const int4*)mrow)[0];
      m1v.v = ((const int4*)mrow)[1];
      #pragma unroll
      for (int r = 0; r < 4; ++r) {
        s[i][r]     = (m0v.i[r] != 0) ? s[i][r]     : NEG_MAX_F32;
        s[i][4 + r] = (m1v.i[r] != 0) ? s[i][4 + r] : NEG_MAX_F32;
      }
    }

    // Online softmax over this 32-key slab (per query = per lane col).
    float mloc = -INFINITY;
    #pragma unroll
    for (int i = 0; i < 2; ++i)
      #pragma unroll
      for (int r = 0; r < 8; ++r) mloc = fmaxf(mloc, s[i][r]);
    mloc = fmaxf(mloc, __shfl_xor(mloc, 16, 32));
    float mnew  = fmaxf(rowMax, mloc);
    float alpha = __expf(rowMax - mnew);
    rowMax = mnew;

    float psum = 0.f;
    FragU pf;  // exp(P) packed directly into the A-fragment this lane owns.
    #pragma unroll
    for (int i = 0; i < 2; ++i)
      #pragma unroll
      for (int r = 0; r < 8; ++r) {
        float p = __expf(s[i][r] - mnew);
        psum += p;
        pf.h[i * 8 + r] = (bf16)p;
      }
    rowSum = alpha * rowSum + (psum + __shfl_xor(psum, 16, 32));

    // Rescale O (per-row alpha broadcast: row q-local = r + 8*hh).
    float af[8];
    #pragma unroll
    for (int r = 0; r < 8; ++r) af[r] = __shfl(alpha, r + 8 * hh, 32);
    #pragma unroll
    for (int c = 0; c < 4; ++c)
      #pragma unroll
      for (int r = 0; r < 8; ++r) o[c][r] *= af[r];

    // O += P*V: A = pf (16q x 32k), B = V^T chunks (32k x 16d).
    #pragma unroll
    for (int c = 0; c < 4; ++c) {
      FragU vf;
      int d = c * 16 + ln;
      vf.u[0] = sVt4[d * 4 + hh];
      vf.u[1] = sVt4[d * 4 + hh + 2];
      o[c] = WMMA_BF16(pf.v, vf.v, o[c]);
    }

#if USE_TDM
    if (wave == 0) __builtin_amdgcn_s_wait_tensorcnt((short)0);  // next K tile done
#endif
    __syncthreads();   // publishes next buffer; retires current reads
  }

  // Normalize and store context bf16 [b, n, h*64 + d].
  float inv = 1.0f / rowSum;
  float nf[8];
  #pragma unroll
  for (int r = 0; r < 8; ++r) nf[r] = __shfl(inv, r + 8 * hh, 32);
  #pragma unroll
  for (int c = 0; c < 4; ++c) {
    int d = c * 16 + ln;
    #pragma unroll
    for (int r = 0; r < 8; ++r) {
      int qg2 = q0 + r + 8 * hh;
      Ctx[((size_t)(bb * 2048 + qg2)) * 1024 + h * 64 + d] = (bf16)(o[c][r] * nf[r]);
    }
  }
}

// ---------------------------------------------------------------- launch ----
extern "C" void kernel_launch(void* const* d_in, const int* in_sizes, int n_in,
                              void* d_out, int out_size, void* d_ws, size_t ws_size,
                              hipStream_t stream) {
  (void)in_sizes; (void)n_in; (void)out_size; (void)ws_size;
  const float* hidden = (const float*)d_in[0];
  const int*   mask   = (const int*)d_in[1];
  const float* Wq = (const float*)d_in[2]; const float* bq = (const float*)d_in[3];
  const float* Wk = (const float*)d_in[4]; const float* bk = (const float*)d_in[5];
  const float* Wv = (const float*)d_in[6]; const float* bv = (const float*)d_in[7];
  const float* Wo = (const float*)d_in[8]; const float* bo = (const float*)d_in[9];

  char* ws = (char*)d_ws;
  const size_t MiB = 1u << 20;
  bf16* Xb   = (bf16*)(ws + 0 * MiB);
  bf16* Wqb  = (bf16*)(ws + 8 * MiB);
  bf16* Wkb  = (bf16*)(ws + 10 * MiB);
  bf16* Wvb  = (bf16*)(ws + 12 * MiB);
  bf16* Wob  = (bf16*)(ws + 14 * MiB);
  bf16* Qb   = (bf16*)(ws + 16 * MiB);
  bf16* Kb   = (bf16*)(ws + 24 * MiB);
  bf16* Vb   = (bf16*)(ws + 32 * MiB);
  bf16* Ctxb = (bf16*)(ws + 40 * MiB);

  // 1) fp32 -> bf16 conversions.
  cast_f32_bf16<<<4096, 256, 0, stream>>>(hidden, Xb, 4096 * 1024 / 4);
  cast_f32_bf16<<<1024, 256, 0, stream>>>(Wq, Wqb, 1024 * 1024 / 4);
  cast_f32_bf16<<<1024, 256, 0, stream>>>(Wk, Wkb, 1024 * 1024 / 4);
  cast_f32_bf16<<<1024, 256, 0, stream>>>(Wv, Wvb, 1024 * 1024 / 4);
  cast_f32_bf16<<<1024, 256, 0, stream>>>(Wo, Wob, 1024 * 1024 / 4);

  // 2) fused QKV projection (z selects Q/K/V), scale folded into Q.
  qkv_gemm<<<dim3(32, 16, 3), 256, 0, stream>>>(Xb, Wqb, Wkb, Wvb, bq, bk, bv,
                                                Qb, Kb, Vb);

  // 3) masked flash attention.
  tree_attn<<<dim3(16, 16, 2), 256, 0, stream>>>(Qb, Kb, Vb, mask, Ctxb);

  // 4) output projection, fp32 + bias into d_out.
  out_gemm<<<dim3(32, 16, 1), 256, 0, stream>>>(Ctxb, Wob, bo, (float*)d_out);
}
